// TGCN_33097017983358
// MI455X (gfx1250) — matrix operand order repaired and checked by reading, not compiled, not run on hardware
//
#include <hip/hip_runtime.h>
#include <hip/hip_bf16.h>
#include <cstddef>

#define BDIM  16
#define NNODE 2048
#define TT    12
#define FIN   2
#define HDIM  64
#define BN    (BDIM * NNODE)   // 32768
#define BT    (BDIM * TT)      // 192
#define NCOL1 (BT * FIN)       // 384
#define NCOL2 (BT * HDIM)      // 12288

typedef __attribute__((ext_vector_type(16))) _Float16 v16h;
typedef __attribute__((ext_vector_type(8)))  _Float16 v8h;
typedef __attribute__((ext_vector_type(8)))  float    v8f;

// ---------------- WMMA fragment helpers (wave32, 16x16x32 f16) ----------------

// A-frag: 16x32 f16 tile from row-major storage, tile origin p0, leading dim lda.
// Lane L<16: row=L, K [0..15]; Lane L>=16: row=L-16, K [16..31] (contiguous halves).
__device__ __forceinline__ v16h load_a_frag(const _Float16* p0, int lda) {
    int lane = threadIdx.x & 31;
    const _Float16* p = p0 + (size_t)(lane & 15) * lda + ((lane & 16) ? 16 : 0);
    v8h lo = *(const v8h*)(p);
    v8h hi = *(const v8h*)(p + 8);
    v16h r;
#pragma unroll
    for (int i = 0; i < 8; ++i) { r[i] = lo[i]; r[i + 8] = hi[i]; }
    return r;
}

// B-frag: 32x16 f16 tile from N-major (transposed) storage Bt[n][k], leading dim ldb.
// Lane L<16: col=L, K [0..15]; Lane L>=16: col=L-16, K [16..31] (contiguous halves).
__device__ __forceinline__ v16h load_b_frag(const _Float16* p0, int ldb) {
    int lane = threadIdx.x & 31;
    const _Float16* p = p0 + (size_t)(lane & 15) * ldb + ((lane & 16) ? 16 : 0);
    return *(const v16h*)(p);
}

__device__ __forceinline__ v8f wmma16(v16h a, v16h b, v8f c) {
    return __builtin_amdgcn_wmma_f32_16x16x32_f16(false, a, false, b, (short)0, c,
                                                  false, false);
}

__device__ __forceinline__ float sigmoidf_(float x) { return 1.f / (1.f + __expf(-x)); }

// Software-pipelined 16x64 (per wave) x K GEMM core. Double-buffers the A/B
// fragments so the loads for step k+32 are in flight during the WMMAs of step k.
// Register budget: 4 accums (32 VGPRs) + 2x40 fragment VGPRs -> no scratch spills.
__device__ __forceinline__ void gemm_core_16x64(const _Float16* __restrict__ Arow, int lda,
                                                const _Float16* __restrict__ Bt, int ldb,
                                                int K, v8f acc[4]) {
    v16h a  = load_a_frag(Arow, lda);
    v16h b0 = load_b_frag(Bt, ldb);
    v16h b1 = load_b_frag(Bt + (size_t)16 * ldb, ldb);
    v16h b2 = load_b_frag(Bt + (size_t)32 * ldb, ldb);
    v16h b3 = load_b_frag(Bt + (size_t)48 * ldb, ldb);
    for (int k = 32; k < K; k += 32) {
        __builtin_prefetch(Arow + k + 64, 0, 3);
        v16h an  = load_a_frag(Arow + k, lda);
        v16h b0n = load_b_frag(Bt + k, ldb);
        v16h b1n = load_b_frag(Bt + (size_t)16 * ldb + k, ldb);
        v16h b2n = load_b_frag(Bt + (size_t)32 * ldb + k, ldb);
        v16h b3n = load_b_frag(Bt + (size_t)48 * ldb + k, ldb);
        acc[0] = wmma16(a, b0, acc[0]);
        acc[1] = wmma16(a, b1, acc[1]);
        acc[2] = wmma16(a, b2, acc[2]);
        acc[3] = wmma16(a, b3, acc[3]);
        a = an; b0 = b0n; b1 = b1n; b2 = b2n; b3 = b3n;
    }
    acc[0] = wmma16(a, b0, acc[0]);
    acc[1] = wmma16(a, b1, acc[1]);
    acc[2] = wmma16(a, b2, acc[2]);
    acc[3] = wmma16(a, b3, acc[3]);
}

// ---------------- Stage 1: BatchNorm stats (per node) ----------------
__global__ void k_bn_stats(const float* __restrict__ X, const float* __restrict__ gamma,
                           const float* __restrict__ beta, float* __restrict__ scale,
                           float* __restrict__ shift) {
    int n = blockIdx.x * blockDim.x + threadIdx.x;
    if (n >= NNODE) return;
    float s = 0.f, ss = 0.f;
    for (int b = 0; b < BDIM; ++b) {
        const float* p = X + (size_t)(b * NNODE + n) * (TT * FIN);
        for (int i = 0; i < TT * FIN; ++i) { float v = p[i]; s += v; ss += v * v; }
    }
    const float inv = 1.f / (float)(BDIM * TT * FIN);
    float mean = s * inv;
    float var  = ss * inv - mean * mean;
    float sc   = gamma[n] * rsqrtf(var + 1e-5f);
    scale[n] = sc;
    shift[n] = beta[n] - mean * sc;
}

// ---------------- f32 -> f16 cast ----------------
__global__ void k_cast_f16(const float* __restrict__ src, _Float16* __restrict__ dst, int n) {
    int i = blockIdx.x * blockDim.x + threadIdx.x;
    if (i < n) dst[i] = (_Float16)src[i];
}

// W2t[h'][h] = W2[h][h']  (64x64 transpose to f16)
__global__ void k_w2t(const float* __restrict__ W2, _Float16* __restrict__ W2t) {
    int i = blockIdx.x * blockDim.x + threadIdx.x;
    if (i >= HDIM * HDIM) return;
    int h = i & 63, hp = i >> 6;
    W2t[i] = (_Float16)W2[h * HDIM + hp];
}

// ---------------- Stage 2: build X1t[j][m] = bn(X[b,m,t,f]) (transposed B operand) --------
__global__ void k_x1t(const float* __restrict__ X, const float* __restrict__ scale,
                      const float* __restrict__ shift, _Float16* __restrict__ X1t) {
    int i = blockIdx.x * blockDim.x + threadIdx.x;
    if (i >= NCOL1 * NNODE) return;
    int m = i & (NNODE - 1);
    int j = i >> 11;                       // (b*TT+t)*FIN+f
    int f = j & 1, t = (j >> 1) % TT, b = j / (TT * FIN);
    float x = X[((size_t)(b * NNODE + m) * TT + t) * FIN + f];
    X1t[(size_t)j * NNODE + m] = (_Float16)(x * scale[m] + shift[m]);
}

// ---------------- Stage 3: AX = A(2048x2048) @ X1(2048x384), f32 out ----------------
__global__ __launch_bounds__(128, 1)
void k_gemm_ax(const _Float16* __restrict__ A, const _Float16* __restrict__ Bt,
               float* __restrict__ C) {
    int w = threadIdx.x >> 5, lane = threadIdx.x & 31;
    int row0 = blockIdx.y * 64 + w * 16;
    int col0 = blockIdx.x * 64;
    v8f acc[4] = {{}, {}, {}, {}};
    gemm_core_16x64(A + (size_t)row0 * NNODE, NNODE,
                    Bt + (size_t)col0 * NNODE, NNODE, NNODE, acc);
    int rl = (lane & 16) ? 8 : 0, cl = lane & 15;
#pragma unroll
    for (int c = 0; c < 4; ++c)
#pragma unroll
        for (int v = 0; v < 8; ++v)
            C[(size_t)(row0 + rl + v) * NCOL1 + col0 + 16 * c + cl] = acc[c][v];
}

// ---------------- Stage 4: G1[m][bt*64+h] = relu(AX*W1 + b1), f16 ----------------
__global__ void k_g1(const float* __restrict__ AX, const float* __restrict__ W1,
                     const float* __restrict__ b1, _Float16* __restrict__ G1) {
    size_t i = (size_t)blockIdx.x * blockDim.x + threadIdx.x;
    if (i >= (size_t)NNODE * NCOL2) return;
    int j = (int)(i % NCOL2), m = (int)(i / NCOL2);
    int h = j & 63, bt = j >> 6;
    float a0 = AX[(size_t)m * NCOL1 + bt * 2];
    float a1 = AX[(size_t)m * NCOL1 + bt * 2 + 1];
    float v  = a0 * W1[h] + a1 * W1[HDIM + h] + b1[h];
    G1[i] = (_Float16)fmaxf(v, 0.f);
}

// ---------------- Stage 5: Zt[bt*64+h'][m] = (G1_bt @ W2)[m][h'], f16 transposed ----------
__global__ __launch_bounds__(128, 1)
void k_z(const _Float16* __restrict__ G1, const _Float16* __restrict__ W2t,
         _Float16* __restrict__ Zt) {
    int w = threadIdx.x >> 5, lane = threadIdx.x & 31;
    int bt = blockIdx.y;
    int row0 = blockIdx.x * 64 + w * 16;
    const _Float16* Ab = G1 + (size_t)bt * HDIM;    // column-block offset within G1 rows
    v8f acc[4] = {{}, {}, {}, {}};
#pragma unroll
    for (int k = 0; k < HDIM; k += 32) {
        v16h a = load_a_frag(Ab + (size_t)row0 * NCOL2 + k, NCOL2);
#pragma unroll
        for (int c = 0; c < 4; ++c) {
            v16h b = load_b_frag(W2t + (16 * c) * HDIM + k, HDIM);
            acc[c] = wmma16(a, b, acc[c]);
        }
    }
    int rl = (lane & 16) ? 8 : 0, cl = lane & 15;
#pragma unroll
    for (int c = 0; c < 4; ++c) {
        int j = bt * HDIM + 16 * c + cl;
        v8h outv;
#pragma unroll
        for (int v = 0; v < 8; ++v) outv[v] = (_Float16)acc[c][v];
        *(v8h*)(Zt + (size_t)j * NNODE + row0 + rl) = outv;   // m-contiguous 16B store
    }
}

// ---------------- Stage 6: seq[t][b*N+m][h] = sigmoid(A@Z + b2), dominant GEMM ----------
__global__ __launch_bounds__(128, 1)
void k_g2(const _Float16* __restrict__ A, const _Float16* __restrict__ Zt,
          const float* __restrict__ b2, _Float16* __restrict__ seq) {
    int w = threadIdx.x >> 5, lane = threadIdx.x & 31;
    int row0 = blockIdx.y * 64 + w * 16;   // node m
    int col0 = blockIdx.x * 64;            // flat (b,t,h) column
    v8f acc[4] = {{}, {}, {}, {}};
    gemm_core_16x64(A + (size_t)row0 * NNODE, NNODE,
                    Zt + (size_t)col0 * NNODE, NNODE, NNODE, acc);
    int rl = (lane & 16) ? 8 : 0, cl = lane & 15;
#pragma unroll
    for (int c = 0; c < 4; ++c) {
        int j = col0 + 16 * c + cl;
        int h = j & 63, bt = j >> 6;
        int b = bt / TT, t = bt % TT;
        float bias = b2[h];
        size_t base = ((size_t)t * BN + (size_t)b * NNODE) * HDIM + h;
#pragma unroll
        for (int v = 0; v < 8; ++v)
            seq[base + (size_t)(row0 + rl + v) * HDIM] =
                (_Float16)sigmoidf_(acc[c][v] + bias);
    }
}

// ---------------- Stage 7: fused GRU step (gx,gh GEMMs + gate math), in-place h ----------
// r/z gates accumulate BOTH x@Wih and h@Whh into one accumulator (WMMA C-chaining);
// only the n gate keeps gxn/ghn separate (needed for r*ghn). 16 accums = 128 VGPRs.
__global__ __launch_bounds__(128, 1)
void k_gru(const _Float16* __restrict__ xin, _Float16* __restrict__ hbuf,
           const _Float16* __restrict__ Wih, const _Float16* __restrict__ Whh,
           const float* __restrict__ bih, const float* __restrict__ bhh,
           _Float16* __restrict__ yout, float* __restrict__ out0) {
    int w = threadIdx.x >> 5, lane = threadIdx.x & 31;
    int row0 = blockIdx.x * 64 + w * 16;
    v8f zero = {};
    v8f accr[4], accz[4], axn[4], ahn[4];
#pragma unroll
    for (int c = 0; c < 4; ++c) { accr[c] = zero; accz[c] = zero; axn[c] = zero; ahn[c] = zero; }
#pragma unroll
    for (int k = 0; k < HDIM; k += 32) {
        v16h a_x = load_a_frag(xin  + (size_t)row0 * HDIM + k, HDIM);
        v16h a_h = load_a_frag(hbuf + (size_t)row0 * HDIM + k, HDIM);
#pragma unroll
        for (int c = 0; c < 4; ++c) {
            accr[c] = wmma16(a_x, load_b_frag(Wih + (size_t)(16 * c) * HDIM + k, HDIM), accr[c]);
            accr[c] = wmma16(a_h, load_b_frag(Whh + (size_t)(16 * c) * HDIM + k, HDIM), accr[c]);
            accz[c] = wmma16(a_x, load_b_frag(Wih + (size_t)(64 + 16 * c) * HDIM + k, HDIM), accz[c]);
            accz[c] = wmma16(a_h, load_b_frag(Whh + (size_t)(64 + 16 * c) * HDIM + k, HDIM), accz[c]);
            axn[c]  = wmma16(a_x, load_b_frag(Wih + (size_t)(128 + 16 * c) * HDIM + k, HDIM), axn[c]);
            ahn[c]  = wmma16(a_h, load_b_frag(Whh + (size_t)(128 + 16 * c) * HDIM + k, HDIM), ahn[c]);
        }
    }
    int rl = (lane & 16) ? 8 : 0, cl = lane & 15;
#pragma unroll
    for (int c = 0; c < 4; ++c) {
        int h = 16 * c + cl;
        float br = bih[h] + bhh[h];
        float bz = bih[64 + h] + bhh[64 + h];
        float bxn = bih[128 + h], bhn = bhh[128 + h];
#pragma unroll
        for (int v = 0; v < 8; ++v) {
            int row = row0 + rl + v;
            float r = sigmoidf_(accr[c][v] + br);
            float z = sigmoidf_(accz[c][v] + bz);
            float n = tanhf(axn[c][v] + bxn + r * (ahn[c][v] + bhn));
            float hold = (float)hbuf[(size_t)row * HDIM + h];
            float hn = (1.f - z) * n + z * hold;
            hbuf[(size_t)row * HDIM + h] = (_Float16)hn;
            yout[(size_t)row * HDIM + h] = (_Float16)hn;
            if (out0) out0[(size_t)row * HDIM + h] = hn;
        }
    }
}

// ---------------- Stage 8: final projection out[b,n,d] = outs0 @ Wl + bl ----------------
__global__ void k_proj(const float* __restrict__ out0, const float* __restrict__ Wl,
                       const float* __restrict__ bl, float* __restrict__ out, int d) {
    int row = blockIdx.x * blockDim.x + threadIdx.x;
    if (row >= BN) return;
    float s = bl[0];
    const float* p = out0 + (size_t)row * HDIM;
#pragma unroll
    for (int h = 0; h < HDIM; ++h) s += p[h] * Wl[h];
    out[(size_t)row * TT + d] = s;
}

// ---------------- host ----------------
extern "C" void kernel_launch(void* const* d_in, const int* in_sizes, int n_in,
                              void* d_out, int out_size, void* d_ws, size_t ws_size,
                              hipStream_t stream) {
    (void)in_sizes; (void)n_in; (void)out_size; (void)ws_size;
    const float* A   = (const float*)d_in[0];
    const float* X   = (const float*)d_in[1];
    const float* gam = (const float*)d_in[2];
    const float* bet = (const float*)d_in[3];
    const float* W1  = (const float*)d_in[4];
    const float* b1  = (const float*)d_in[5];
    const float* W2  = (const float*)d_in[6];
    const float* b2  = (const float*)d_in[7];
    const float* Wih = (const float*)d_in[8];
    const float* Whh = (const float*)d_in[9];
    const float* bih = (const float*)d_in[10];
    const float* bhh = (const float*)d_in[11];
    const float* Wl  = (const float*)d_in[12];
    const float* bl  = (const float*)d_in[13];
    float* out = (float*)d_out;

    char* p = (char*)d_ws;
    auto carve = [&](size_t bytes) -> char* {
        char* r = p; p += (bytes + 255) & ~(size_t)255; return r;
    };
    _Float16* Af16  = (_Float16*)carve((size_t)NNODE * NNODE * 2);
    float*    scale = (float*)carve(NNODE * 4);
    float*    shift = (float*)carve(NNODE * 4);
    _Float16* X1t   = (_Float16*)carve((size_t)NCOL1 * NNODE * 2);
    float*    AX    = (float*)carve((size_t)NNODE * NCOL1 * 4);
    _Float16* Zt    = (_Float16*)carve((size_t)NCOL2 * NNODE * 2);
    _Float16* W2t   = (_Float16*)carve(HDIM * HDIM * 2);
    _Float16* Wih16 = (_Float16*)carve(3 * HDIM * HDIM * 2);
    _Float16* Whh16 = (_Float16*)carve(3 * HDIM * HDIM * 2);
    _Float16* seqA  = (_Float16*)carve((size_t)TT * BN * HDIM * 2);
    _Float16* seqB  = (_Float16*)carve((size_t)TT * BN * HDIM * 2); // aliased with G1
    _Float16* hbuf  = (_Float16*)carve((size_t)BN * HDIM * 2);
    float*    out0  = (float*)carve((size_t)BN * HDIM * 4);
    _Float16* G1 = seqB;   // NNODE*NCOL2 == TT*BN*HDIM exactly; G1 dead before seqB written

    // preprocessing
    k_bn_stats<<<(NNODE + 127) / 128, 128, 0, stream>>>(X, gam, bet, scale, shift);
    k_cast_f16<<<(NNODE * NNODE + 255) / 256, 256, 0, stream>>>(A, Af16, NNODE * NNODE);
    k_cast_f16<<<(3 * HDIM * HDIM + 255) / 256, 256, 0, stream>>>(Wih, Wih16, 3 * HDIM * HDIM);
    k_cast_f16<<<(3 * HDIM * HDIM + 255) / 256, 256, 0, stream>>>(Whh, Whh16, 3 * HDIM * HDIM);
    k_w2t<<<(HDIM * HDIM + 255) / 256, 256, 0, stream>>>(W2, W2t);
    k_x1t<<<(NCOL1 * NNODE) / 256, 256, 0, stream>>>(X, scale, shift, X1t);

    // GCN layer 1: AX = A @ bn(X); g1 = relu(AX @ W1 + b1)
    dim3 g3(NCOL1 / 64, NNODE / 64);
    k_gemm_ax<<<g3, 128, 0, stream>>>(Af16, X1t, AX);
    size_t ng1 = (size_t)NNODE * NCOL2;
    k_g1<<<(unsigned)((ng1 + 255) / 256), 256, 0, stream>>>(AX, W1, b1, G1);

    // GCN layer 2: Z = g1 @ W2 (per bt); seqA = sigmoid(A @ Z + b2) in GRU layout
    dim3 g5(NNODE / 64, BT);
    k_z<<<g5, 128, 0, stream>>>(G1, W2t, Zt);
    dim3 g6(NCOL2 / 64, NNODE / 64);
    k_g2<<<g6, 128, 0, stream>>>(Af16, Zt, b2, seqA);

    // GRU encoder
    hipMemsetAsync(hbuf, 0, (size_t)BN * HDIM * 2, stream);
    const size_t SSTEP = (size_t)BN * HDIM;
    for (int t = 0; t < TT; ++t)
        k_gru<<<BN / 64, 128, 0, stream>>>(seqA + t * SSTEP, hbuf, Wih16, Whh16,
                                           bih, bhh, seqB + t * SSTEP, nullptr);
    // GRU decoder: re-scan full sequence T_OUT times, emit t[0] each iteration
    _Float16* cur = seqB;
    _Float16* nxt = seqA;
    for (int d = 0; d < TT; ++d) {
        for (int t = 0; t < TT; ++t)
            k_gru<<<BN / 64, 128, 0, stream>>>(cur + t * SSTEP, hbuf, Wih16, Whh16,
                                               bih, bhh, nxt + t * SSTEP,
                                               (t == 0) ? out0 : nullptr);
        k_proj<<<BN / 256, 256, 0, stream>>>(out0, Wl, bl, out, d);
        _Float16* tmp = cur; cur = nxt; nxt = tmp;
    }
}
